// MTGE_39934605919046
// MI455X (gfx1250) — compile-verified, matrix operand order checked by reading, not run on hardware
//
#include <hip/hip_runtime.h>
#include <hip/hip_bf16.h>
#include <math.h>

#define B_ROWS 2048
#define D_DIM  512
#define L_HIST 128
#define LDA 40                 // LDS row stride (halves): 80B, keeps 16B alignment
#define LDB 40
#define AS_STRIDE (128 * LDA)  // halves per A buffer
#define BS_STRIDE (128 * LDB)  // halves per B buffer

typedef __attribute__((ext_vector_type(16))) _Float16 v16h;
typedef __attribute__((ext_vector_type(8)))  _Float16 v8h;
typedef __attribute__((ext_vector_type(8)))  float    v8f;

union V16U { v16h v; v8h h[2]; };

__device__ inline v8h cvt8(float4 x, float4 y) {
    v8h h = { (_Float16)x.x, (_Float16)x.y, (_Float16)x.z, (_Float16)x.w,
              (_Float16)y.x, (_Float16)y.y, (_Float16)y.z, (_Float16)y.w };
    return h;
}

// ---------------------------------------------------------------------------
// embeds_u = (e1*e^-4 + e2*e^-3 + e3*e^-2 + e4*e^-1) / S
// ---------------------------------------------------------------------------
__global__ void fuse_u_kernel(const float* __restrict__ e1, const float* __restrict__ e2,
                              const float* __restrict__ e3, const float* __restrict__ e4,
                              float* __restrict__ out, int n) {
    int i = blockIdx.x * blockDim.x + threadIdx.x;
    if (i < n) {
        const float w0 = 0.018315638889f, w1 = 0.049787068368f;
        const float w2 = 0.135335283237f, w3 = 0.367879441171f;
        const float invS = 1.0f / 0.571317431664f;
        out[i] = (e1[i]*w0 + e2[i]*w1 + e3[i]*w2 + e4[i]*w3) * invS;
    }
}

// ---------------------------------------------------------------------------
// d_min[b] = sqrt(min over valid l of ||v_embed[hist[b,l]] - v_embed[nodes[b]]||^2)
// ---------------------------------------------------------------------------
__global__ void dist_kernel(const float* __restrict__ v_embed, const int* __restrict__ nodes_v,
                            const int* __restrict__ hist, const int* __restrict__ hist_len,
                            float* __restrict__ d_min) {
    __shared__ __align__(16) float newv[D_DIM];
    __shared__ float wmin[8];
    const int b   = blockIdx.x;
    const int tid = threadIdx.x;
    const int node = nodes_v[b];
    newv[tid]       = v_embed[(size_t)node * D_DIM + tid];
    newv[tid + 256] = v_embed[(size_t)node * D_DIM + tid + 256];
    __syncthreads();
    const int wave = tid >> 5, lane = tid & 31;
    const int hl = hist_len[b];
    const float4* nv4 = (const float4*)newv;
    float lmin = 3.402823466e+38f;
    for (int l = wave; l < L_HIST; l += 8) {
        if (l >= hl) continue;                       // uniform per wave
        const int idx = hist[b * L_HIST + l];
        const float4* row = (const float4*)(v_embed + (size_t)idx * D_DIM);
        int ln = l + 8;
        if (ln < hl) {
            int idxn = hist[b * L_HIST + ln];
            __builtin_prefetch(v_embed + (size_t)idxn * D_DIM + lane * 16, 0, 0);
        }
        float s = 0.f;
        #pragma unroll
        for (int c = 0; c < 4; ++c) {
            float4 rv = row[c * 32 + lane];
            float4 nv = nv4[c * 32 + lane];
            float dx = rv.x - nv.x, dy = rv.y - nv.y;
            float dz = rv.z - nv.z, dw = rv.w - nv.w;
            s += dx*dx + dy*dy + dz*dz + dw*dw;
        }
        #pragma unroll
        for (int off = 16; off >= 1; off >>= 1) s += __shfl_xor(s, off, 32);
        lmin = fminf(lmin, s);
    }
    if (lane == 0) wmin[wave] = lmin;
    __syncthreads();
    if (tid == 0) {
        float mn = wmin[0];
        #pragma unroll
        for (int w = 1; w < 8; ++w) mn = fminf(mn, wmin[w]);
        d_min[b] = sqrtf(mn);
    }
}

// ---------------------------------------------------------------------------
// c[b] = (||e1-e2|| + ||e2-e3|| + ||e3-e4||) / 3 ; one wave per row
// ---------------------------------------------------------------------------
__global__ void cons_kernel(const float* __restrict__ e1, const float* __restrict__ e2,
                            const float* __restrict__ e3, const float* __restrict__ e4,
                            float* __restrict__ cbuf) {
    const int wave = threadIdx.x >> 5, lane = threadIdx.x & 31;
    const int b = blockIdx.x * 8 + wave;
    const size_t base = (size_t)b * D_DIM;
    float s12 = 0.f, s23 = 0.f, s34 = 0.f;
    #pragma unroll
    for (int j = 0; j < 16; ++j) {
        int k = j * 32 + lane;
        float a1 = e1[base + k], a2 = e2[base + k], a3 = e3[base + k], a4 = e4[base + k];
        float d12 = a1 - a2, d23 = a2 - a3, d34 = a3 - a4;
        s12 += d12 * d12; s23 += d23 * d23; s34 += d34 * d34;
    }
    #pragma unroll
    for (int off = 16; off >= 1; off >>= 1) {
        s12 += __shfl_xor(s12, off, 32);
        s23 += __shfl_xor(s23, off, 32);
        s34 += __shfl_xor(s34, off, 32);
    }
    if (lane == 0) cbuf[b] = (sqrtf(s12) + sqrtf(s23) + sqrtf(s34)) * (1.0f / 3.0f);
}

// ---------------------------------------------------------------------------
// min/max of d_min and c -> scal[0..3]
// ---------------------------------------------------------------------------
__global__ void minmax_kernel(const float* __restrict__ a, const float* __restrict__ c,
                              float* __restrict__ scal) {
    __shared__ float smm[4][32];
    const int tid = threadIdx.x;
    float amn = 3.402823466e+38f, amx = -3.402823466e+38f;
    float cmn = 3.402823466e+38f, cmx = -3.402823466e+38f;
    for (int i = tid; i < B_ROWS; i += 1024) {
        float av = a[i], cv = c[i];
        amn = fminf(amn, av); amx = fmaxf(amx, av);
        cmn = fminf(cmn, cv); cmx = fmaxf(cmx, cv);
    }
    #pragma unroll
    for (int off = 16; off >= 1; off >>= 1) {
        amn = fminf(amn, __shfl_xor(amn, off, 32)); amx = fmaxf(amx, __shfl_xor(amx, off, 32));
        cmn = fminf(cmn, __shfl_xor(cmn, off, 32)); cmx = fmaxf(cmx, __shfl_xor(cmx, off, 32));
    }
    const int wave = tid >> 5, lane = tid & 31;
    if (lane == 0) { smm[0][wave] = amn; smm[1][wave] = amx; smm[2][wave] = cmn; smm[3][wave] = cmx; }
    __syncthreads();
    if (wave == 0) {
        amn = smm[0][lane]; amx = smm[1][lane]; cmn = smm[2][lane]; cmx = smm[3][lane];
        #pragma unroll
        for (int off = 16; off >= 1; off >>= 1) {
            amn = fminf(amn, __shfl_xor(amn, off, 32)); amx = fmaxf(amx, __shfl_xor(amx, off, 32));
            cmn = fminf(cmn, __shfl_xor(cmn, off, 32)); cmx = fmaxf(cmx, __shfl_xor(cmx, off, 32));
        }
        if (lane == 0) { scal[0] = amn; scal[1] = amx; scal[2] = cmn; scal[3] = cmx; }
    }
}

// ---------------------------------------------------------------------------
// Y[m, ncol_off+n] = epilogue( X[m,:K] . W[n,:K] + bias[n] )   (torch Linear)
// Block tile 128(M) x 128(N); 8 waves as 2(M) x 4(N); each wave 64x32 patch
// = 4x2 WMMA tiles of v_wmma_f32_16x16x32_f16. K stepped by 32 through
// double-buffered LDS (fp32->f16 on the fill), next tile prefetched into
// registers before the WMMAs so global loads overlap matrix math; one
// barrier per K-step.
// ---------------------------------------------------------------------------
template <bool GUARD>
__global__ __launch_bounds__(256) void gemm_kernel(
        const float* __restrict__ X, int K,
        const float* __restrict__ W, int Ndim,
        const float* __restrict__ bias,
        const float* __restrict__ bn_g, const float* __restrict__ bn_b,
        const float* __restrict__ bn_m, const float* __restrict__ bn_v,
        int bn_relu, float* __restrict__ Y, int ldc, int ncol_off) {
    __shared__ __align__(16) _Float16 As[2 * AS_STRIDE];
    __shared__ __align__(16) _Float16 Bs[2 * BS_STRIDE];
    const int tid = threadIdx.x;
    const int m0 = blockIdx.y * 128;
    const int n0 = blockIdx.x * 128;
    const int wave = tid >> 5, lane = tid & 31;
    const int wm = wave >> 2, wn = wave & 3;     // 2 waves along M, 4 along N
    const int m16 = lane & 15;
    const int kb  = (lane >> 4) * 8;             // K half-offset within 32-K tile

    // staging: chunk c in [0,512): row = c>>2 (0..127), 8-float group = c&3
    // thread handles chunks tid and tid+256 for both A and B
    const float* pa[2];
    const float* pb[2];
    bool bval[2];
    #pragma unroll
    for (int i = 0; i < 2; ++i) {
        int c = tid + i * 256;
        int row = c >> 2, c8 = c & 3;
        pa[i] = X + (size_t)(m0 + row) * K + c8 * 8;
        int gn = n0 + row;
        bval[i] = !GUARD || (gn < Ndim);
        int gnc = GUARD ? (gn < Ndim ? gn : (Ndim - 1)) : gn;
        pb[i] = W + (size_t)gnc * K + c8 * 8;
    }

    float4 ra[2][2], rb[2][2];
    const float4 f4z = { 0.f, 0.f, 0.f, 0.f };
    #pragma unroll
    for (int i = 0; i < 2; ++i) {
        ra[i][0] = *(const float4*)pa[i];
        ra[i][1] = *(const float4*)(pa[i] + 4);
        pa[i] += 32;
        if (bval[i]) {
            rb[i][0] = *(const float4*)pb[i];
            rb[i][1] = *(const float4*)(pb[i] + 4);
            pb[i] += 32;
        } else { rb[i][0] = f4z; rb[i][1] = f4z; }
    }

    v8f acc[4][2] = {};
    const int nk = K >> 5;
    for (int it = 0; it < nk; ++it) {
        const int buf = it & 1;
        _Float16* as = As + buf * AS_STRIDE;
        _Float16* bs = Bs + buf * BS_STRIDE;
        // convert + store current tile (one b128 store each)
        #pragma unroll
        for (int i = 0; i < 2; ++i) {
            int c = tid + i * 256;
            int row = c >> 2, c8 = c & 3;
            *(v8h*)(as + row * LDA + c8 * 8) = cvt8(ra[i][0], ra[i][1]);
            *(v8h*)(bs + row * LDB + c8 * 8) = cvt8(rb[i][0], rb[i][1]);
        }
        __syncthreads();
        // prefetch next tile into registers (overlaps with WMMAs below)
        if (it + 1 < nk) {
            #pragma unroll
            for (int i = 0; i < 2; ++i) {
                ra[i][0] = *(const float4*)pa[i];
                ra[i][1] = *(const float4*)(pa[i] + 4);
                pa[i] += 32;
                if (bval[i]) {
                    rb[i][0] = *(const float4*)pb[i];
                    rb[i][1] = *(const float4*)(pb[i] + 4);
                    pb[i] += 32;
                } else { rb[i][0] = f4z; rb[i][1] = f4z; }
            }
        }
        // fragments (ISA 7.12.2 16-bit A/B layout) + 8 WMMAs
        V16U af[4], bf[2];
        #pragma unroll
        for (int i = 0; i < 4; ++i) {
            const _Float16* p = as + (wm * 64 + i * 16 + m16) * LDA + kb;
            af[i].h[0] = *(const v8h*)p;
            af[i].h[1] = *(const v8h*)(p + 16);
        }
        #pragma unroll
        for (int j = 0; j < 2; ++j) {
            const _Float16* p = bs + (wn * 32 + j * 16 + m16) * LDB + kb;
            bf[j].h[0] = *(const v8h*)p;
            bf[j].h[1] = *(const v8h*)(p + 16);
        }
        #pragma unroll
        for (int i = 0; i < 4; ++i)
            #pragma unroll
            for (int j = 0; j < 2; ++j)
                acc[i][j] = __builtin_amdgcn_wmma_f32_16x16x32_f16(
                    false, af[i].v, false, bf[j].v, (short)0, acc[i][j], false, false);
        // no trailing barrier: double-buffered LDS
    }

    // epilogue: bias (+ BN eval + ReLU), fp32 store with column offset
    const int rhi = (lane >> 4) * 8;
    #pragma unroll
    for (int i = 0; i < 4; ++i) {
        #pragma unroll
        for (int j = 0; j < 2; ++j) {
            int gn = n0 + wn * 32 + j * 16 + m16;
            if (!GUARD || gn < Ndim) {
                float bia = bias[gn];
                float g = 1.f, bb = 0.f, mu = 0.f, rs = 1.f;
                if (bn_relu) {
                    g  = bn_g[gn]; bb = bn_b[gn];
                    mu = bn_m[gn]; rs = rsqrtf(bn_v[gn] + 1e-5f);
                }
                #pragma unroll
                for (int r = 0; r < 8; ++r) {
                    int gm = m0 + wm * 64 + i * 16 + rhi + r;
                    float v = acc[i][j][r] + bia;
                    if (bn_relu) v = fmaxf((v - mu) * rs * g + bb, 0.f);
                    Y[(size_t)gm * ldc + ncol_off + gn] = v;
                }
            }
        }
    }
}

// ---------------------------------------------------------------------------
// scores[b] = h4[b,:16].w3 + b3 ; out = scores + 6t e^{-6t} * c_n
// ---------------------------------------------------------------------------
__global__ void final_kernel(const float* __restrict__ h4, const float* __restrict__ w3,
                             const float* __restrict__ b3, const float* __restrict__ dmin,
                             const float* __restrict__ cbuf, const float* __restrict__ scal,
                             float* __restrict__ out) {
    int b = blockIdx.x * blockDim.x + threadIdx.x;
    if (b >= B_ROWS) return;
    float s = b3[0];
    #pragma unroll
    for (int j = 0; j < 16; ++j) s += h4[b * 16 + j] * w3[j];
    float t  = (dmin[b] - scal[0]) / (scal[1] - scal[0]);
    float un = 6.f * t * __expf(-6.f * t);
    float cn = (cbuf[b] - scal[2]) / (scal[3] - scal[2]);
    out[b] = s + un * cn;
}

// ---------------------------------------------------------------------------
extern "C" void kernel_launch(void* const* d_in, const int* in_sizes, int n_in,
                              void* d_out, int out_size, void* d_ws, size_t ws_size,
                              hipStream_t stream) {
    const float* v_embed  = (const float*)d_in[0];
    const float* e1       = (const float*)d_in[1];
    const float* e2       = (const float*)d_in[2];
    const float* e3       = (const float*)d_in[3];
    const float* e4       = (const float*)d_in[4];
    const float* embeds_v = (const float*)d_in[5];
    const int*   nodes_v  = (const int*)d_in[6];
    const int*   hist     = (const int*)d_in[7];
    const int*   hist_len = (const int*)d_in[8];
    const float* w_ur1_w = (const float*)d_in[9];  const float* w_ur1_b = (const float*)d_in[10];
    const float* w_ur2_w = (const float*)d_in[11]; const float* w_ur2_b = (const float*)d_in[12];
    const float* w_vr1_w = (const float*)d_in[13]; const float* w_vr1_b = (const float*)d_in[14];
    const float* w_vr2_w = (const float*)d_in[15]; const float* w_vr2_b = (const float*)d_in[16];
    const float* w_uv1_w = (const float*)d_in[17]; const float* w_uv1_b = (const float*)d_in[18];
    const float* w_uv2_w = (const float*)d_in[19]; const float* w_uv2_b = (const float*)d_in[20];
    const float* w_uv3_w = (const float*)d_in[21]; const float* w_uv3_b = (const float*)d_in[22];
    const float *bn1_g = (const float*)d_in[23], *bn1_b = (const float*)d_in[24],
                *bn1_m = (const float*)d_in[25], *bn1_v = (const float*)d_in[26];
    const float *bn2_g = (const float*)d_in[27], *bn2_b = (const float*)d_in[28],
                *bn2_m = (const float*)d_in[29], *bn2_v = (const float*)d_in[30];
    const float *bn3_g = (const float*)d_in[31], *bn3_b = (const float*)d_in[32],
                *bn3_m = (const float*)d_in[33], *bn3_v = (const float*)d_in[34];
    const float *bn4_g = (const float*)d_in[35], *bn4_b = (const float*)d_in[36],
                *bn4_m = (const float*)d_in[37], *bn4_v = (const float*)d_in[38];

    // workspace layout (fp32): ~21 MB total
    float* ws       = (float*)d_ws;
    float* embeds_u = ws;                        // 2048*512
    float* h        = embeds_u + 2048 * 512;     // 2048*512 (x_u pre / x_v pre)
    float* xcat     = h + 2048 * 512;            // 2048*1024 (x_u | x_v)
    float* h3       = xcat + 2048 * 1024;        // 2048*512
    float* h4       = h3 + 2048 * 512;           // 2048*16
    float* dmin     = h4 + 2048 * 16;            // 2048
    float* cbuf     = dmin + 2048;               // 2048
    float* scal     = cbuf + 2048;               // 4

    fuse_u_kernel<<<4096, 256, 0, stream>>>(e1, e2, e3, e4, embeds_u, 2048 * 512);
    dist_kernel<<<2048, 256, 0, stream>>>(v_embed, nodes_v, hist, hist_len, dmin);
    cons_kernel<<<256, 256, 0, stream>>>(e1, e2, e3, e4, cbuf);
    minmax_kernel<<<1, 1024, 0, stream>>>(dmin, cbuf, scal);

    dim3 blk(256);
    dim3 g512(4, 16);   // N=512: 4 col-tiles x 16 row-tiles (128x128 blocks)
    dim3 g16(1, 16);    // N=16
    // x_u tower
    gemm_kernel<false><<<g512, blk, 0, stream>>>(embeds_u, 512, w_ur1_w, 512, w_ur1_b,
                                                 bn1_g, bn1_b, bn1_m, bn1_v, 1, h, 512, 0);
    gemm_kernel<false><<<g512, blk, 0, stream>>>(h, 512, w_ur2_w, 512, w_ur2_b,
                                                 bn1_g, bn1_b, bn1_m, bn1_v, 0, xcat, 1024, 0);
    // x_v tower
    gemm_kernel<false><<<g512, blk, 0, stream>>>(embeds_v, 512, w_vr1_w, 512, w_vr1_b,
                                                 bn2_g, bn2_b, bn2_m, bn2_v, 1, h, 512, 0);
    gemm_kernel<false><<<g512, blk, 0, stream>>>(h, 512, w_vr2_w, 512, w_vr2_b,
                                                 bn2_g, bn2_b, bn2_m, bn2_v, 0, xcat, 1024, 512);
    // joint tower
    gemm_kernel<false><<<g512, blk, 0, stream>>>(xcat, 1024, w_uv1_w, 512, w_uv1_b,
                                                 bn3_g, bn3_b, bn3_m, bn3_v, 1, h3, 512, 0);
    gemm_kernel<true><<<g16, blk, 0, stream>>>(h3, 512, w_uv2_w, 16, w_uv2_b,
                                               bn4_g, bn4_b, bn4_m, bn4_v, 1, h4, 16, 0);
    final_kernel<<<8, 256, 0, stream>>>(h4, w_uv3_w, w_uv3_b, dmin, cbuf, scal, (float*)d_out);
}
